// LinearProjection_7662221656339
// MI455X (gfx1250) — matrix operand order repaired and checked by reading, not compiled, not run on hardware
//
#include <hip/hip_runtime.h>
#include <math.h>

// Sizes from the reference.
#define BB 8
#define NQ 128
#define NK 128
#define DD 256
#define L0S 512
#define L1S 256

typedef __attribute__((ext_vector_type(16))) __bf16 v16bf;
typedef __attribute__((ext_vector_type(8)))  float  v8f;
typedef __attribute__((ext_vector_type(4)))  int    v4i_t;

#if __has_builtin(__builtin_amdgcn_global_load_async_to_lds_b128)
#define USE_ASYNC 1
#else
#define USE_ASYNC 0
#endif

__device__ __forceinline__ void async_cp16(const uint4* src, uint4* dst)
{
#if USE_ASYNC
    __builtin_amdgcn_global_load_async_to_lds_b128(
        (__attribute__((address_space(1))) v4i_t*)src,
        (__attribute__((address_space(3))) v4i_t*)dst, 0, 0);
#else
    *dst = *src;
#endif
}

__device__ __forceinline__ void async_wait0()
{
#if USE_ASYNC
#if __has_builtin(__builtin_amdgcn_s_wait_asynccnt)
    __builtin_amdgcn_s_wait_asynccnt(0);
#else
    asm volatile("s_wait_asynccnt 0x0" ::: "memory");
#endif
#endif
}

// ---------------------------------------------------------------------------
// Kernel 1: aq' = q @ W1[:D] + b1 ; ak = k @ W1[D:]   (fp32, [B*128, 512])
// ---------------------------------------------------------------------------
__global__ void __launch_bounds__(256)
proj_kernel(const float* __restrict__ q, const float* __restrict__ kk,
            const float* __restrict__ W1, const float* __restrict__ b1,
            float* __restrict__ aq, float* __restrict__ ak)
{
    __shared__ float xs[16 * 256];
    const int tid   = threadIdx.x;
    const int grp   = blockIdx.x;        // 0..127
    const int which = grp >> 6;          // 0 => aq (from q), 1 => ak (from k)
    const int r0    = (grp & 63) * 16;   // first flat row (b*128+n)
    const float* X  = which ? kk : q;
    const int koff  = which ? 256 : 0;
    float* dst      = which ? ak : aq;

    const float4* src4 = (const float4*)(X + r0 * 256);
    float4* xs4 = (float4*)xs;
#pragma unroll
    for (int i = 0; i < 4; ++i) xs4[tid + i * 256] = src4[tid + i * 256];
    __syncthreads();

    float acc0[16], acc1[16];
    const int f0 = tid, f1 = tid + 256;
#pragma unroll
    for (int r = 0; r < 16; ++r) {
        acc0[r] = which ? 0.f : b1[f0];
        acc1[r] = which ? 0.f : b1[f1];
    }
    for (int d = 0; d < 256; ++d) {
        const float w0 = W1[(koff + d) * L0S + f0];   // coalesced over tid
        const float w1 = W1[(koff + d) * L0S + f1];
#pragma unroll
        for (int r = 0; r < 16; ++r) {
            const float x = xs[r * 256 + d];
            acc0[r] += x * w0;
            acc1[r] += x * w1;
        }
    }
#pragma unroll
    for (int r = 0; r < 16; ++r) {
        dst[(r0 + r) * L0S + f0] = acc0[r];
        dst[(r0 + r) * L0S + f1] = acc1[r];
    }
}

// ---------------------------------------------------------------------------
// Kernel 2: W2 [512,256] fp32 row-major -> W2T bf16 column-major [256][512]
// ---------------------------------------------------------------------------
__global__ void __launch_bounds__(256)
w2t_kernel(const float* __restrict__ W2, unsigned short* __restrict__ w2t)
{
    const int idx = blockIdx.x * 256 + threadIdx.x;  // 0..131071
    const int kk  = idx >> 8;                        // K index 0..511
    const int n   = idx & 255;                       // N index 0..255
    union { __bf16 h; unsigned short u; } cv;
    cv.h = (__bf16)W2[idx];
    w2t[n * L0S + kk] = cv.u;
}

// ---------------------------------------------------------------------------
// Kernel 3 (dominant): scores via WMMA bf16, fragments built on the fly.
// Block: 256 threads = 8 waves. WG tile: 16 q x 16 k. Wave: 2 q x all 16 k.
// Accumulators: acc[2 q][4 n] v8f = 64 VGPRs.
// LDS (dynamic, 192 KB): aq 16x512 f32 | ak 16x512 f32 | 2x W2 strip (64 KB)
// W2 strips are double-buffered: async-load strip ng+1 while computing ng.
// ---------------------------------------------------------------------------
__global__ void __launch_bounds__(256, 1)
pair_mlp_scores(const float* __restrict__ aqG, const float* __restrict__ akG,
                const unsigned short* __restrict__ w2t,
                const float* __restrict__ b2, const float* __restrict__ wo,
                float* __restrict__ scoresG)
{
    extern __shared__ char smem[];
    float* aqs = (float*)smem;                                // 32 KB
    float* aks = aqs + 16 * L0S;                              // 32 KB
    unsigned short* w2s0 = (unsigned short*)(aks + 16 * L0S); // 64 KB
    unsigned short* w2s1 = w2s0 + 64 * L0S;                   // 64 KB

    const int tid = threadIdx.x;
    const int kb = blockIdx.x, qb = blockIdx.y, b = blockIdx.z;

    // ---- async-stage aq' (16 rows), ak (16 rows) and W2 strip 0 ----
    {
        const uint4* s4 = (const uint4*)(aqG + (b * NQ + qb * 16) * L0S);
        uint4* d4 = (uint4*)aqs;
#pragma unroll
        for (int i = 0; i < 8; ++i) async_cp16(s4 + tid + i * 256, d4 + tid + i * 256);
        const uint4* s4b = (const uint4*)(akG + (b * NK + kb * 16) * L0S);
        uint4* d4b = (uint4*)aks;
#pragma unroll
        for (int i = 0; i < 8; ++i) async_cp16(s4b + tid + i * 256, d4b + tid + i * 256);
        const uint4* s4w = (const uint4*)(w2t);
        uint4* d4w = (uint4*)w2s0;
#pragma unroll
        for (int i = 0; i < 16; ++i) async_cp16(s4w + tid + i * 256, d4w + tid + i * 256);
    }
    async_wait0();
    __syncthreads();

    const int lane = tid & 31;
    const int wave = tid >> 5;
    const int q0   = wave * 2;          // local q base (2 q's per wave)
    const int l16  = lane & 15;
    const int hi   = lane >> 4;         // lane group (K-half select)

    float part[2][8];                   // fused score partials [qi][C-row]
#pragma unroll
    for (int qi = 0; qi < 2; ++qi)
#pragma unroll
        for (int r = 0; r < 8; ++r) part[qi][r] = 0.f;

    for (int ng = 0; ng < 4; ++ng) {    // 4 feature groups of 64 N each
        const int nbase = ng * 64;
        const unsigned short* cur = (ng & 1) ? w2s1 : w2s0;

        // kick off async copy of the NEXT strip into the other buffer;
        // it completes under the compute below (hidden copy latency).
        if (ng < 3) {
            unsigned short* nxt = (ng & 1) ? w2s0 : w2s1;
            const uint4* s = (const uint4*)(w2t + (nbase + 64) * L0S);
            uint4* d = (uint4*)nxt;
#pragma unroll
            for (int i = 0; i < 16; ++i) async_cp16(s + tid + i * 256, d + tid + i * 256);
        }

        v8f acc[2][4];
#pragma unroll
        for (int qi = 0; qi < 2; ++qi)
#pragma unroll
            for (int nt = 0; nt < 4; ++nt) acc[qi][nt] = (v8f)0.f;

        for (int ks = 0; ks < 16; ++ks) {     // K steps of 32
            // B fragments: lane holds K 8*hi..+7 and 16+8*hi..+7 of column (nt*16+l16)
            v16bf bfrag[4];
#pragma unroll
            for (int nt = 0; nt < 4; ++nt) {
                const int c = nt * 16 + l16;
                const uint4* p = (const uint4*)(cur + c * L0S + ks * 32 + hi * 8);
                union { uint4 u[2]; v16bf v; } cv;
                cv.u[0] = p[0];
                cv.u[1] = p[2];
                bfrag[nt] = cv.v;
            }
            // per-lane ak chunks for this K step (shared across the 2 q's)
            const float* akr = aks + l16 * L0S;
            const int off1 = ks * 32 + hi * 8;
            const int off2 = off1 + 16;
            float akv0[8], akv1[8];
#pragma unroll
            for (int j = 0; j < 8; ++j) { akv0[j] = akr[off1 + j]; akv1[j] = akr[off2 + j]; }

#pragma unroll
            for (int qi = 0; qi < 2; ++qi) {
                const float* aqr = aqs + (q0 + qi) * L0S;
                v16bf afrag;
#pragma unroll
                for (int j = 0; j < 8; ++j) {
                    float h0 = aqr[off1 + j] + akv0[j]; h0 = h0 > 0.f ? h0 : 0.f;
                    float h1 = aqr[off2 + j] + akv1[j]; h1 = h1 > 0.f ? h1 : 0.f;
                    afrag[j]     = (__bf16)h0;
                    afrag[8 + j] = (__bf16)h1;
                }
#pragma unroll
                for (int nt = 0; nt < 4; ++nt)
                    acc[qi][nt] = __builtin_amdgcn_wmma_f32_16x16x32_bf16(
                        false, afrag, false, bfrag[nt],
                        (short)0, acc[qi][nt], false, false);
            }
        }

        // fused layer-2 epilogue: part += relu(h2 + b2) * Wo
#pragma unroll
        for (int qi = 0; qi < 2; ++qi)
#pragma unroll
            for (int nt = 0; nt < 4; ++nt) {
                const int n = nbase + nt * 16 + l16;
                const float b2n = b2[n];
                const float won = wo[n];
#pragma unroll
                for (int r = 0; r < 8; ++r) {
                    float vv = acc[qi][nt][r] + b2n;
                    vv = vv > 0.f ? vv : 0.f;
                    part[qi][r] += vv * won;
                }
            }

        // next strip's async writes done + visible to all waves
        async_wait0();
        __syncthreads();
    }

    // reduce over N within each 16-lane half (ds_swizzle path), then store
#pragma unroll
    for (int qi = 0; qi < 2; ++qi)
#pragma unroll
        for (int r = 0; r < 8; ++r) {
            float s = part[qi][r];
#pragma unroll
            for (int m = 1; m < 16; m <<= 1) s += __shfl_xor(s, m, 32);
            part[qi][r] = s;
        }
    if (l16 == 0) {  // lane 0 holds rows 0..7, lane 16 holds rows 8..15
#pragma unroll
        for (int qi = 0; qi < 2; ++qi) {
            const int qg = qb * 16 + q0 + qi;
#pragma unroll
            for (int r = 0; r < 8; ++r) {
                const int kl = hi * 8 + r;
                scoresG[(b * NQ + qg) * NK + kb * 16 + kl] = part[qi][r];
            }
        }
    }
}

// ---------------------------------------------------------------------------
// Kernel 4: per-(b,q) softmax over k (mask -> -inf; +bo dropped: softmax is
// shift-invariant) then out = weights @ v.   Block = 128 threads.
// ---------------------------------------------------------------------------
__global__ void __launch_bounds__(128)
softmax_av(const float* __restrict__ scores, const unsigned char* __restrict__ mask,
           const float* __restrict__ v, float* __restrict__ out)
{
    __shared__ float red[128];
    __shared__ float sw[128];
    const int tid = threadIdx.x;
    const int b  = blockIdx.x >> 7;
    const int qv = blockIdx.x & 127;
    const int base = (b * NQ + qv) * NK;

    float s = scores[base + tid];
    if (mask[base + tid]) s = -__builtin_inff();

    red[tid] = s; __syncthreads();
    for (int off = 64; off > 0; off >>= 1) {
        if (tid < off) red[tid] = fmaxf(red[tid], red[tid + off]);
        __syncthreads();
    }
    const float mx = red[0]; __syncthreads();
    const float e = __expf(s - mx);
    red[tid] = e; __syncthreads();
    for (int off = 64; off > 0; off >>= 1) {
        if (tid < off) red[tid] += red[tid + off];
        __syncthreads();
    }
    const float inv = 1.f / red[0]; __syncthreads();
    sw[tid] = e * inv; __syncthreads();

    float acc0 = 0.f, acc1 = 0.f;
    const float* vb = v + b * NK * DD;
    for (int kx = 0; kx < NK; ++kx) {
        const float w = sw[kx];
        acc0 += w * vb[kx * DD + tid];          // coalesced over tid
        acc1 += w * vb[kx * DD + tid + 128];
    }
    float* ob = out + (b * NQ + qv) * DD;
    ob[tid]       = acc0;
    ob[tid + 128] = acc1;
}

// ---------------------------------------------------------------------------
// Launch
// ---------------------------------------------------------------------------
extern "C" void kernel_launch(void* const* d_in, const int* in_sizes, int n_in,
                              void* d_out, int out_size, void* d_ws, size_t ws_size,
                              hipStream_t stream)
{
    const float* q  = (const float*)d_in[0];
    const float* k  = (const float*)d_in[1];
    const float* v  = (const float*)d_in[2];
    const unsigned char* mask = (const unsigned char*)d_in[3];
    const float* W1 = (const float*)d_in[4];
    const float* b1 = (const float*)d_in[5];
    const float* W2 = (const float*)d_in[6];
    const float* b2 = (const float*)d_in[7];
    const float* Wo = (const float*)d_in[8];
    // d_in[9] = bo: uniform shift before softmax -> no effect on output.

    char* ws = (char*)d_ws;
    float*          aq     = (float*)(ws);                               // 2 MB
    float*          ak     = (float*)(ws + (2u << 20));                  // 2 MB
    unsigned short* w2t    = (unsigned short*)(ws + (4u << 20));         // 256 KB
    float*          scores = (float*)(ws + (4u << 20) + (256u << 10));   // 512 KB

    proj_kernel<<<128, 256, 0, stream>>>(q, k, W1, b1, aq, ak);
    w2t_kernel<<<512, 256, 0, stream>>>(W2, w2t);

    dim3 gB(NK / 16, NQ / 16, BB);   // 8 x 8 x 8 = 512 workgroups
    pair_mlp_scores<<<gB, 256, 192 * 1024, stream>>>(aq, ak, w2t, b2, Wo, scores);

    softmax_av<<<BB * NQ, 128, 0, stream>>>(scores, mask, v, (float*)d_out);
}
